// Llama_76287209111758
// MI455X (gfx1250) — matrix operand order repaired and checked by reading, not compiled, not run on hardware
//
#include <hip/hip_runtime.h>
#include <math.h>

// ---------------------------------------------------------------------------
// Model constants (from the reference)
// ---------------------------------------------------------------------------
#define LAY 6
#define DIM 768
#define IDIM 768
#define NH 8
#define NKV 4
#define HD 96
#define VOC 50263
#define SEQ 1280
#define BAT 2
#define BT (BAT * SEQ)          // 2560 activation rows
#define HALFHD (HD / 2)         // 48 rope pairs
#define RMS_EPS 1e-6f

typedef __attribute__((ext_vector_type(16))) _Float16 v16h;
typedef __attribute__((ext_vector_type(8)))  _Float16 v8h;
typedef __attribute__((ext_vector_type(8)))  float    v8f;

// ---------------------------------------------------------------------------
// f16 WMMA GEMM.  C[M,N] = A[M,K] @ B^T  where:
//   A  : f16 row-major [M][K]            (activations, pre-converted)
//   Bt : f16 row-major [N][K]            (weights pre-transposed+converted,
//                                         or naturally [N,K]: K-rows, V^T)
// Output modes: 0 = f32 (+bias, +C residual), 1 = f16, 2 = f16 transposed.
// Block tile 128x128x32, 8 waves (wave32), per-wave 2x4 grid of
// v_wmma_f32_16x16x32_f16 (32x64 patch).
// Staging is a raw 16B copy -> GLOBAL_LOAD_ASYNC_TO_LDS_B128 (ASYNCcnt),
// fragments are contiguous ds_load_b128 reads.
// Assumptions (hold at every call site): M % 128 == 0, K % 32 == 0,
// A/B rows 16B-aligned.  Only N may be ragged (guarded path).
// ---------------------------------------------------------------------------
#define BM 128
#define BN 128
#define BK 32
#define AKP 40    // As row stride (f16): 80B/row -> 16B-aligned segments
#define BKP 40    // Bs row stride (f16)

__device__ __forceinline__ void async_cp_b128(const _Float16* g, const _Float16* l) {
    // VDST = LDS byte address (low 32 bits of flat LDS pointer), VADDR = global
    unsigned loff = (unsigned)(unsigned long long)l;
    asm volatile("global_load_async_to_lds_b128 %0, %1, off"
                 :: "v"(loff), "v"(g) : "memory");
}
__device__ __forceinline__ void wait_async() {
    asm volatile("s_wait_asynccnt 0x0" ::: "memory");
}

__global__ __launch_bounds__(256)
void gemm_f16_wmma(const _Float16* __restrict__ A, int lda,
                   const _Float16* __restrict__ Bt, int ldb,
                   void* __restrict__ Cv, int ldc,
                   const float* __restrict__ bias, int addC, int outMode,
                   int M, int N, int Nstage, int K) {
    __shared__ _Float16 As[BM * AKP];   // 10 KB
    __shared__ _Float16 Bs[BN * BKP];   // 10 KB  (Bs[n][k])

    const int tid   = threadIdx.x;
    const int lane  = tid & 31;
    const int wave  = tid >> 5;
    const int waveM = wave & 3;         // 4 waves along M
    const int waveN = wave >> 2;        // 2 waves along N
    const int rowBlk = blockIdx.y * BM;
    const int colBlk = blockIdx.x * BN;
    const int lh  = lane >> 4;
    const int l16 = lane & 15;
    const bool fastB = (colBlk + BN) <= Nstage;   // uniform across block

    v8f acc[2][4];
#pragma unroll
    for (int mt = 0; mt < 2; ++mt)
#pragma unroll
        for (int nt = 0; nt < 4; ++nt)
            acc[mt][nt] = (v8f){};

    for (int k0 = 0; k0 < K; k0 += BK) {
        // ---- stage A tile (128 rows x 32 halves): 2 async 16B copies/thread
#pragma unroll
        for (int i = 0; i < 2; ++i) {
            int seg = i * 256 + tid;            // 512 segments of 16B
            int r = seg >> 2, s = (seg & 3) << 3;
            async_cp_b128(&A[(long)(rowBlk + r) * lda + k0 + s],
                          &As[r * AKP + s]);
        }
        // ---- stage B tile (128 n-rows x 32 halves), same pattern ----
        if (fastB) {
#pragma unroll
            for (int i = 0; i < 2; ++i) {
                int seg = i * 256 + tid;
                int r = seg >> 2, s = (seg & 3) << 3;
                async_cp_b128(&Bt[(long)(colBlk + r) * ldb + k0 + s],
                              &Bs[r * BKP + s]);
            }
        } else {
#pragma unroll 1
            for (int i = 0; i < 16; ++i) {      // ragged N tail (vocab)
                int flat = i * 256 + tid;
                int c = flat >> 5, r = flat & 31;
                int gn = colBlk + c;
                _Float16 v = (gn < N) ? Bt[(long)gn * ldb + k0 + r]
                                      : (_Float16)0.0f;
                Bs[c * BKP + r] = v;
            }
        }
        wait_async();
        __syncthreads();

        // ---- A fragments: 16-bit A 16x32 layout (ISA 7.12.2) ----
        v16h af[2];
#pragma unroll
        for (int mt = 0; mt < 2; ++mt) {
            int row = waveM * 32 + mt * 16 + l16;
            const v8h a0 = *(const v8h*)&As[row * AKP + lh * 8];
            const v8h a1 = *(const v8h*)&As[row * AKP + 16 + lh * 8];
            af[mt] = __builtin_shufflevector(a0, a1,
                      0, 1, 2, 3, 4, 5, 6, 7, 8, 9, 10, 11, 12, 13, 14, 15);
        }
        // ---- B fragments: col fixed per lane, k = lh*8 + e / 16 + lh*8 + e (matches A) ----
        v16h bf[4];
#pragma unroll
        for (int nt = 0; nt < 4; ++nt) {
            int col = waveN * 64 + nt * 16 + l16;
            const v8h b0 = *(const v8h*)&Bs[col * BKP + lh * 8];
            const v8h b1 = *(const v8h*)&Bs[col * BKP + 16 + lh * 8];
            bf[nt] = __builtin_shufflevector(b0, b1,
                      0, 1, 2, 3, 4, 5, 6, 7, 8, 9, 10, 11, 12, 13, 14, 15);
        }
        // ---- 8 WMMAs per wave per K-step ----
#pragma unroll
        for (int mt = 0; mt < 2; ++mt)
#pragma unroll
            for (int nt = 0; nt < 4; ++nt)
                acc[mt][nt] = __builtin_amdgcn_wmma_f32_16x16x32_f16(
                    false, af[mt], false, bf[nt],
                    (short)0, acc[mt][nt], false, false);
        __syncthreads();
    }

    // ---- epilogue: C/D layout M = vgpr + 8*(lane/16), N = lane%16 ----
    if (outMode == 0) {
        float* C = (float*)Cv;
#pragma unroll
        for (int mt = 0; mt < 2; ++mt)
#pragma unroll
            for (int nt = 0; nt < 4; ++nt)
#pragma unroll
                for (int j = 0; j < 8; ++j) {
                    int row = rowBlk + waveM * 32 + mt * 16 + lh * 8 + j;
                    int col = colBlk + waveN * 64 + nt * 16 + l16;
                    if (col < N) {
                        float v = acc[mt][nt][j];
                        if (bias) v += bias[col];
                        if (addC) v += C[(long)row * ldc + col];
                        C[(long)row * ldc + col] = v;
                    }
                }
    } else if (outMode == 1) {
        _Float16* C = (_Float16*)Cv;
#pragma unroll
        for (int mt = 0; mt < 2; ++mt)
#pragma unroll
            for (int nt = 0; nt < 4; ++nt)
#pragma unroll
                for (int j = 0; j < 8; ++j) {
                    int row = rowBlk + waveM * 32 + mt * 16 + lh * 8 + j;
                    int col = colBlk + waveN * 64 + nt * 16 + l16;
                    if (col < N)
                        C[(long)row * ldc + col] = (_Float16)acc[mt][nt][j];
                }
    } else {  // f16 transposed store: C[col][row]  (V^T production)
        _Float16* C = (_Float16*)Cv;
#pragma unroll
        for (int mt = 0; mt < 2; ++mt)
#pragma unroll
            for (int nt = 0; nt < 4; ++nt)
#pragma unroll
                for (int j = 0; j < 8; ++j) {
                    int row = rowBlk + waveM * 32 + mt * 16 + lh * 8 + j;
                    int col = colBlk + waveN * 64 + nt * 16 + l16;
                    if (col < N)
                        C[(long)col * ldc + row] = (_Float16)acc[mt][nt][j];
                }
    }
}

// ---------------------------------------------------------------------------
// Convert + transpose: in f32 [R][Cn] -> out f16 [Cn][R]   (weights -> [N][K])
// ---------------------------------------------------------------------------
__global__ __launch_bounds__(256)
void convT_kernel(const float* __restrict__ in, _Float16* __restrict__ out,
                  int R, int Cn) {
    __shared__ float tile[32][33];
    const int bc = blockIdx.x * 32;   // input col  / output row
    const int br = blockIdx.y * 32;   // input row  / output col
    const int tx = threadIdx.x & 31;
    const int ty = threadIdx.x >> 5;  // 0..7
#pragma unroll
    for (int i = 0; i < 32; i += 8) {
        int r = br + ty + i, c = bc + tx;
        tile[ty + i][tx] = (r < R && c < Cn) ? in[(long)r * Cn + c] : 0.0f;
    }
    __syncthreads();
#pragma unroll
    for (int i = 0; i < 32; i += 8) {
        int oc = bc + ty + i;   // output row (= input col)
        int orr = br + tx;      // output col (= input row)
        if (oc < Cn && orr < R)
            out[(long)oc * R + orr] = (_Float16)tile[tx][ty + i];
    }
}

// ---------------------------------------------------------------------------
// Embedding gather: h[b,t,:] = emb[x[b,t],:]  (f32 residual stream)
// ---------------------------------------------------------------------------
__global__ void embed_kernel(const int* __restrict__ x,
                             const float* __restrict__ emb,
                             float* __restrict__ h, int total) {
    int i = blockIdx.x * blockDim.x + threadIdx.x;
    if (i >= total) return;
    int row = i / DIM, c = i % DIM;
    h[i] = emb[(long)x[row] * DIM + c];
}

// ---------------------------------------------------------------------------
// RMSNorm (f32 in -> f16 out), one 256-thread block per row of D=768.
// ---------------------------------------------------------------------------
__global__ __launch_bounds__(256)
void rmsnorm_kernel(const float* __restrict__ in, const float* __restrict__ w,
                    _Float16* __restrict__ out) {
    __shared__ float red[256];
    const long base = (long)blockIdx.x * DIM;
    const int tid = threadIdx.x;
    float p = 0.0f;
    for (int i = tid; i < DIM; i += 256) { float v = in[base + i]; p += v * v; }
    red[tid] = p; __syncthreads();
    for (int s = 128; s > 0; s >>= 1) {
        if (tid < s) red[tid] += red[tid + s];
        __syncthreads();
    }
    float rstd = rsqrtf(red[0] / (float)DIM + RMS_EPS);
    for (int i = tid; i < DIM; i += 256)
        out[base + i] = (_Float16)(in[base + i] * rstd * w[i]);
}

// ---------------------------------------------------------------------------
// RoPE (interleaved pairs): f32 src -> f16 dst, (B,T,nh,HD) layout.
// ---------------------------------------------------------------------------
__global__ void rope_kernel(const float* __restrict__ src,
                            _Float16* __restrict__ dst,
                            const float* __restrict__ cosT,
                            const float* __restrict__ sinT,
                            int nh, int total) {
    int idx = blockIdx.x * blockDim.x + threadIdx.x;
    if (idx >= total) return;
    int pair = idx % HALFHD;
    int rest = idx / HALFHD;
    int head = rest % nh;
    int bt   = rest / nh;
    int t    = bt % SEQ;
    long base = ((long)bt * nh + head) * HD + 2 * pair;
    float c = cosT[t * HALFHD + pair];
    float s = sinT[t * HALFHD + pair];
    float x1 = src[base], x2 = src[base + 1];
    dst[base]     = (_Float16)(x1 * c - x2 * s);
    dst[base + 1] = (_Float16)(x1 * s + x2 * c);
}

// ---------------------------------------------------------------------------
// Masked softmax over one head's scores (T x T): f32 in -> f16 probs out.
// Reference form: s' = 2*s*inv (k<=q)  |  s*inv - 1e9 (k>q)
// ---------------------------------------------------------------------------
__global__ __launch_bounds__(256)
void softmax_kernel(const float* __restrict__ sc, _Float16* __restrict__ p,
                    float inv) {
    __shared__ float row[SEQ];
    __shared__ float red[256];
    const int q = blockIdx.x;
    const int tid = threadIdx.x;
    float mx = -3.0e38f;
    for (int k = tid; k < SEQ; k += 256) {
        float s = sc[(long)q * SEQ + k] * inv;
        s = (k <= q) ? (2.0f * s) : (s - 1.0e9f);
        row[k] = s;
        mx = fmaxf(mx, s);
    }
    red[tid] = mx; __syncthreads();
    for (int s = 128; s > 0; s >>= 1) {
        if (tid < s) red[tid] = fmaxf(red[tid], red[tid + s]);
        __syncthreads();
    }
    mx = red[0]; __syncthreads();
    float sum = 0.0f;
    for (int k = tid; k < SEQ; k += 256) {
        float e = __expf(row[k] - mx);
        row[k] = e; sum += e;
    }
    red[tid] = sum; __syncthreads();
    for (int s = 128; s > 0; s >>= 1) {
        if (tid < s) red[tid] += red[tid + s];
        __syncthreads();
    }
    float rcp = 1.0f / red[0];
    for (int k = tid; k < SEQ; k += 256)
        p[(long)q * SEQ + k] = (_Float16)(row[k] * rcp);
}

// ---------------------------------------------------------------------------
// o = silu(g) * u   (f32 in -> f16 out)
// ---------------------------------------------------------------------------
__global__ void silu_mul_kernel(const float* __restrict__ g,
                                const float* __restrict__ u,
                                _Float16* __restrict__ o, int n) {
    int i = blockIdx.x * blockDim.x + threadIdx.x;
    if (i >= n) return;
    float x = g[i];
    o[i] = (_Float16)((x / (1.0f + __expf(-x))) * u[i]);
}

// ---------------------------------------------------------------------------
// Host orchestration
// ---------------------------------------------------------------------------
static inline void launch_gemm(const _Float16* A, int lda,
                               const _Float16* Bt, int ldb,
                               void* C, int ldc,
                               const float* bias, int addC, int outMode,
                               int M, int N, int Nstage, int K,
                               hipStream_t stream) {
    dim3 grid((N + BN - 1) / BN, (M + BM - 1) / BM);
    gemm_f16_wmma<<<grid, 256, 0, stream>>>(A, lda, Bt, ldb, C, ldc,
                                            bias, addC, outMode,
                                            M, N, Nstage, K);
}
static inline void launch_convT(const float* in, _Float16* out, int R, int Cn,
                                hipStream_t stream) {
    dim3 grid((Cn + 31) / 32, (R + 31) / 32);
    convT_kernel<<<grid, 256, 0, stream>>>(in, out, R, Cn);
}

extern "C" void kernel_launch(void* const* d_in, const int* in_sizes, int n_in,
                              void* d_out, int out_size, void* d_ws, size_t ws_size,
                              hipStream_t stream) {
    // inputs in setup_inputs() order
    const int*   x       = (const int*)  d_in[0];
    // d_in[1] = mask (causality applied analytically)
    const float* emb     = (const float*)d_in[2];
    const float* rms_w   = (const float*)d_in[3];
    const float* wq      = (const float*)d_in[4];
    const float* wk      = (const float*)d_in[5];
    const float* wv      = (const float*)d_in[6];
    const float* wo      = (const float*)d_in[7];
    const float* gate_w  = (const float*)d_in[8];
    const float* gate_b  = (const float*)d_in[9];
    const float* up_w    = (const float*)d_in[10];
    const float* up_b    = (const float*)d_in[11];
    const float* down_w  = (const float*)d_in[12];
    const float* down_b  = (const float*)d_in[13];
    const float* frms_w  = (const float*)d_in[14];
    const float* fgate_w = (const float*)d_in[15];
    const float* fup_w   = (const float*)d_in[16];
    const float* fdown_w = (const float*)d_in[17];
    const float* proj_w  = (const float*)d_in[18];
    const float* cosT    = (const float*)d_in[19];
    const float* sinT    = (const float*)d_in[20];
    float* out = (float*)d_out;

    // ---------------- workspace layout (requires ws_size >= ~200 MB) -------
    // f32 region
    float* wsf = (float*)d_ws;
    const long SZ_H  = (long)BT * DIM;            // 1,966,080
    const long SZ_KV = (long)BT * NKV * HD;
    const long SZ_SC = (long)SEQ * SEQ;
    float* h   = wsf;  wsf += SZ_H;               // residual stream
    float* qf  = wsf;  wsf += SZ_H;               // q pre-rope (f32)
    float* kf  = wsf;  wsf += SZ_KV;              // k pre-rope (f32)
    float* sc  = wsf;  wsf += SZ_SC;              // one head's scores
    float* g1  = wsf;  wsf += (long)BT * IDIM;
    float* g2  = wsf;  wsf += (long)BT * IDIM;
    // f16 region
    _Float16* wsh = (_Float16*)wsf;
    const long WQS = (long)DIM * NH * HD;         // 589,824
    const long WKS = (long)DIM * NKV * HD;        // 294,912
    const long WMS = (long)DIM * IDIM;            // 589,824
    const long PERL = WQS + 2 * WKS + WMS * 4;    // per-layer weight halves
    _Float16* wl16   = wsh;  wsh += PERL * LAY;   // per-layer weights [N][K]
    _Float16* fg16   = wsh;  wsh += WMS;
    _Float16* fu16   = wsh;  wsh += WMS;
    _Float16* fd16   = wsh;  wsh += WMS;
    _Float16* proj16 = wsh;  wsh += (long)VOC * DIM;   // [VOC][DIM]
    _Float16* xn16   = wsh;  wsh += SZ_H;
    _Float16* q16    = wsh;  wsh += SZ_H;
    _Float16* k16    = wsh;  wsh += SZ_KV;
    _Float16* vT16   = wsh;  wsh += (long)512 * BT;    // V^T padded to 512 rows
    _Float16* p16    = wsh;  wsh += SZ_SC;
    _Float16* ao16   = wsh;  wsh += SZ_H;
    _Float16* g16    = wsh;  wsh += (long)BT * IDIM;
    _Float16* hf16   = wsh;  wsh += SZ_H;

    const float inv_sqrt = 0.10206207261596577f;  // 1/sqrt(96)

    // ---------------- one-time (per call) weight convert+transpose ---------
    _Float16* lw[LAY][7];
    for (int l = 0; l < LAY; ++l) {
        _Float16* p = wl16 + (long)l * PERL;
        lw[l][0] = p;               p += WQS;   // wq^T  [768][768]
        lw[l][1] = p;               p += WKS;   // wk^T  [384][768]
        lw[l][2] = p;               p += WKS;   // wv^T  [384][768]
        lw[l][3] = p;               p += WQS;   // wo^T  [768][768]
        lw[l][4] = p;               p += WMS;   // gate^T
        lw[l][5] = p;               p += WMS;   // up^T
        lw[l][6] = p;               p += WMS;   // down^T
        launch_convT(wq + (long)l * WQS, lw[l][0], DIM, NH * HD, stream);
        launch_convT(wk + (long)l * WKS, lw[l][1], DIM, NKV * HD, stream);
        launch_convT(wv + (long)l * WKS, lw[l][2], DIM, NKV * HD, stream);
        launch_convT(wo + (long)l * WQS, lw[l][3], NH * HD, DIM, stream);
        launch_convT(gate_w + (long)l * WMS, lw[l][4], DIM, IDIM, stream);
        launch_convT(up_w   + (long)l * WMS, lw[l][5], DIM, IDIM, stream);
        launch_convT(down_w + (long)l * WMS, lw[l][6], IDIM, DIM, stream);
    }
    launch_convT(fgate_w, fg16, DIM, IDIM, stream);
    launch_convT(fup_w,   fu16, DIM, IDIM, stream);
    launch_convT(fdown_w, fd16, IDIM, DIM, stream);
    launch_convT(proj_w,  proj16, DIM, VOC, stream);

    // ---------------- embedding -------------------------------------------
    {
        int total = BT * DIM;
        embed_kernel<<<(total + 255) / 256, 256, 0, stream>>>(x, emb, h, total);
    }

    // ---------------- transformer layers ----------------------------------
    for (int l = 0; l < LAY; ++l) {
        const float* rw = rms_w  + (long)l * DIM;
        const float* gb = gate_b + (long)l * IDIM;
        const float* ub = up_b   + (long)l * IDIM;
        const float* db = down_b + (long)l * DIM;

        rmsnorm_kernel<<<BT, 256, 0, stream>>>(h, rw, xn16);

        // QKV projections (WMMA): q,k -> f32 (rope); v -> f16 transposed V^T
        launch_gemm(xn16, DIM, lw[l][0], DIM, qf, NH * HD, nullptr, 0, 0,
                    BT, NH * HD, NH * HD, DIM, stream);
        launch_gemm(xn16, DIM, lw[l][1], DIM, kf, NKV * HD, nullptr, 0, 0,
                    BT, NKV * HD, NKV * HD, DIM, stream);
        launch_gemm(xn16, DIM, lw[l][2], DIM, vT16, BT, nullptr, 0, 2,
                    BT, NKV * HD, NKV * HD, DIM, stream);

        // RoPE: f32 -> f16
        {
            int tq = BT * NH  * HALFHD;
            int tk = BT * NKV * HALFHD;
            rope_kernel<<<(tq + 255) / 256, 256, 0, stream>>>(qf, q16, cosT, sinT, NH,  tq);
            rope_kernel<<<(tk + 255) / 256, 256, 0, stream>>>(kf, k16, cosT, sinT, NKV, tk);
        }

        // attention per (b, head); GQA: head hh -> kv head hh/2
        for (int b = 0; b < BAT; ++b) {
            for (int hh = 0; hh < NH; ++hh) {
                int hk = hh >> 1;
                const _Float16* qh = q16 + (long)b * SEQ * NH  * HD + (long)hh * HD;
                const _Float16* kh = k16 + (long)b * SEQ * NKV * HD + (long)hk * HD;
                const _Float16* vh = vT16 + (long)(hk * HD) * BT + (long)b * SEQ;
                _Float16* aoh      = ao16 + (long)b * SEQ * NH  * HD + (long)hh * HD;

                // scores = q @ k^T  (K=HD; B naturally [N=T][K=HD])
                launch_gemm(qh, NH * HD, kh, NKV * HD, sc, SEQ,
                            nullptr, 0, 0, SEQ, SEQ, SEQ, HD, stream);
                softmax_kernel<<<SEQ, 256, 0, stream>>>(sc, p16, inv_sqrt);
                // ao = attn @ v  (B = V^T [N=HD][K=T], rows padded to 128)
                launch_gemm(p16, SEQ, vh, BT, aoh, NH * HD,
                            nullptr, 0, 1, SEQ, HD, BN, SEQ, stream);
            }
        }

        // output projection with residual: h += ao @ wo
        launch_gemm(ao16, NH * HD, lw[l][3], NH * HD, h, DIM,
                    nullptr, 1, 0, BT, DIM, DIM, NH * HD, stream);

        // MLP block
        rmsnorm_kernel<<<BT, 256, 0, stream>>>(h, rw, xn16);
        launch_gemm(xn16, DIM, lw[l][4], DIM, g1, IDIM, gb, 0, 0,
                    BT, IDIM, IDIM, DIM, stream);
        launch_gemm(xn16, DIM, lw[l][5], DIM, g2, IDIM, ub, 0, 0,
                    BT, IDIM, IDIM, DIM, stream);
        {
            int n = BT * IDIM;
            silu_mul_kernel<<<(n + 255) / 256, 256, 0, stream>>>(g1, g2, g16, n);
        }
        launch_gemm(g16, IDIM, lw[l][6], IDIM, h, DIM, db, 1, 0,
                    BT, DIM, DIM, IDIM, stream);
    }

    // ---------------- final norm + final MLP + vocab projection ------------
    rmsnorm_kernel<<<BT, 256, 0, stream>>>(h, frms_w, xn16);
    launch_gemm(xn16, DIM, fg16, DIM, g1, IDIM, nullptr, 0, 0,
                BT, IDIM, IDIM, DIM, stream);
    launch_gemm(xn16, DIM, fu16, DIM, g2, IDIM, nullptr, 0, 0,
                BT, IDIM, IDIM, DIM, stream);
    {
        int n = BT * IDIM;
        silu_mul_kernel<<<(n + 255) / 256, 256, 0, stream>>>(g1, g2, g16, n);
    }
    launch_gemm(g16, IDIM, fd16, IDIM, hf16, DIM, nullptr, 0, 1,
                BT, DIM, DIM, IDIM, stream);
    // logits (N = 50263: last column block takes the guarded staging path)
    launch_gemm(hf16, DIM, proj16, DIM, out, VOC, nullptr, 0, 0,
                BT, VOC, VOC, DIM, stream);
}
